// SchNetInteractionBlock_15333033246866
// MI455X (gfx1250) — matrix operand order repaired and checked by reading, not compiled, not run on hardware
//
#include <hip/hip_runtime.h>

#define FDIM 128
#define RBAS 20
#define BN   8000
#define NP   512000

typedef __attribute__((ext_vector_type(16))) __bf16 v16bf;
typedef __attribute__((ext_vector_type(8)))  float  v8f;

struct __align__(16) U128 { unsigned int a, b, c, d; };

union Frag {
    v16bf  v;
    U128   u[2];
    __bf16 bf[16];
};

__device__ __forceinline__ v8f vzero() {
    v8f z = {0.f, 0.f, 0.f, 0.f, 0.f, 0.f, 0.f, 0.f};
    return z;
}

__device__ __forceinline__ v8f wmma_bf16(v16bf a, v16bf b, v8f c) {
    // D = A(16x32 bf16) * B(32x16 bf16) + C(16x16 f32)
    return __builtin_amdgcn_wmma_f32_16x16x32_bf16(false, a, false, b, (short)0, c, false, false);
}

// ShiftedSoftplus: softplus(x) - log(2), numerically stable
__device__ __forceinline__ float sspf(float x) {
    return fmaxf(x, 0.f) + log1pf(__expf(-fabsf(x))) - 0.6931471805599453f;
}

// ---- Fragment loaders (layouts per CDNA5 ISA 7.12.2) ----
// A 16x32 bf16: lane m = L&15, hf = L>>4.
//   elems 0..7  -> K = kb + hf*8 + i
//   elems 8..15 -> K = kb + 16 + hf*8 + i
__device__ __forceinline__ v16bf load_afrag_f32(const float* row, int hf, int kb) {
    Frag f;
    const float* r0 = row + kb + hf * 8;
    const float* r1 = row + kb + 16 + hf * 8;
#pragma unroll
    for (int i = 0; i < 8; i++) {
        f.bf[i]     = (__bf16)r0[i];
        f.bf[8 + i] = (__bf16)r1[i];
    }
    return f.v;
}

__device__ __forceinline__ v16bf load_afrag_lds(const __bf16* row, int hf, int kb) {
    Frag f;
    f.u[0] = *reinterpret_cast<const U128*>(row + kb + hf * 8);
    f.u[1] = *reinterpret_cast<const U128*>(row + kb + 16 + hf * 8);
    return f.v;
}

// B 32x16 bf16: lane n-col = L&15, hf = L>>4; elems 0..7 -> K = kb + hf*8 + i, elems 8..15 -> K = kb + 16 + hf*8 + i
// wt is the [N][K] transposed bf16 panel, Kt = row length.
__device__ __forceinline__ v16bf load_bfrag(const __bf16* wt, int Kt, int n, int hf, int kb) {
    Frag f;
    const __bf16* p = wt + (size_t)n * Kt + kb + hf * 8;
    f.u[0] = *reinterpret_cast<const U128*>(p);
    f.u[1] = *reinterpret_cast<const U128*>(p + 16);
    return f.v;
}

// ======================= prep: zero agg, transpose/convert weights =======================
__global__ void schnet_prep(const float* W_in, const float* Wf1, const float* Wf2,
                            const float* Wo1, const float* Wo2,
                            float* agg, __bf16* Wt_in, __bf16* Wf2t,
                            __bf16* Wo1t, __bf16* Wo2t, __bf16* Wf1t) {
    int tid = blockIdx.x * blockDim.x + threadIdx.x;
    int stride = gridDim.x * blockDim.x;
    for (int i = tid; i < BN * FDIM; i += stride) agg[i] = 0.f;
    for (int i = tid; i < FDIM * FDIM; i += stride) {
        int n = i / FDIM, k = i % FDIM;
        Wt_in[i] = (__bf16)W_in[k * FDIM + n];
        Wf2t[i]  = (__bf16)Wf2[k * FDIM + n];
        Wo1t[i]  = (__bf16)Wo1[k * FDIM + n];
        Wo2t[i]  = (__bf16)Wo2[k * FDIM + n];
    }
    for (int i = tid; i < FDIM * 32; i += stride) {
        int n = i / 32, k = i % 32;
        Wf1t[i] = (k < RBAS) ? (__bf16)Wf1[k * FDIM + n] : (__bf16)0.f;
    }
}

// ======================= h = x @ W_in + b_in =======================
#define AT_TILES 4
__global__ void __launch_bounds__(256) schnet_atom_in(const float* x, const __bf16* Wt,
                                                      const float* b_in, float* h) {
    const int lane = threadIdx.x & 31;
    const int wave = threadIdx.x >> 5;
    const int hf   = lane >> 4;
    const int n    = wave * 16 + (lane & 15);
    const int m    = lane & 15;

    v16bf bfr[4];
#pragma unroll
    for (int s = 0; s < 4; s++) bfr[s] = load_bfrag(Wt, FDIM, n, hf, s * 32);
    const float bias = b_in[n];

    for (int t = 0; t < AT_TILES; t++) {
        const int row0 = (blockIdx.x * AT_TILES + t) * 16;
        const float* arow = x + (size_t)(row0 + m) * FDIM;
        v8f acc = vzero();
#pragma unroll
        for (int s = 0; s < 4; s++)
            acc = wmma_bf16(load_afrag_f32(arow, hf, s * 32), bfr[s], acc);
#pragma unroll
        for (int v = 0; v < 8; v++) {
            int rm = v + 8 * hf;
            h[(size_t)(row0 + rm) * FDIM + n] = acc[v] + bias;
        }
    }
}

// ======================= fused pair kernel =======================
// Wij = (ssp(f_ij@Wf1+bf1)@Wf2 + bf2)*rcut ; scatter-add h[idx_j]*Wij into agg[idx_i]
#define PAIR_TILES 8
__global__ void __launch_bounds__(256) schnet_pair(const float* f_ij, const int* idx_i,
                                                   const int* idx_j, const float* rcut,
                                                   const __bf16* Wf1t, const float* bf1,
                                                   const __bf16* Wf2t, const float* bf2,
                                                   const float* h, float* agg) {
    __shared__ __align__(16) __bf16 tS[16][136];
    const int lane = threadIdx.x & 31;
    const int wave = threadIdx.x >> 5;
    const int hf   = lane >> 4;
    const int n    = wave * 16 + (lane & 15);
    const int m    = lane & 15;

    // register-resident B fragments for both GEMMs
    v16bf b1 = load_bfrag(Wf1t, 32, n, hf, 0);
    v16bf b2[4];
#pragma unroll
    for (int s = 0; s < 4; s++) b2[s] = load_bfrag(Wf2t, FDIM, n, hf, s * 32);
    const float bias1 = bf1[n];
    const float bias2 = bf2[n];

    for (int t = 0; t < PAIR_TILES; t++) {
        const int p0 = (blockIdx.x * PAIR_TILES + t) * 16;

        // GEMM1: A = f_ij rows, K padded 20 -> 32
        Frag fa;
        const float* fr = f_ij + (size_t)(p0 + m) * RBAS;
#pragma unroll
        for (int i = 0; i < 8; i++) fa.bf[i] = (__bf16)fr[hf * 8 + i];
#pragma unroll
        for (int i = 0; i < 8; i++) {
            int k = 16 + hf * 8 + i;
            fa.bf[8 + i] = (k < RBAS) ? (__bf16)fr[k] : (__bf16)0.f;
        }
        v8f c1 = wmma_bf16(fa.v, b1, vzero());

        __syncthreads();  // protect tS against previous iteration's readers
#pragma unroll
        for (int v = 0; v < 8; v++)
            tS[v + 8 * hf][n] = (__bf16)sspf(c1[v] + bias1);
        __syncthreads();

        // GEMM2: K = 128
        v8f c2 = vzero();
#pragma unroll
        for (int s = 0; s < 4; s++)
            c2 = wmma_bf16(load_afrag_lds(&tS[m][0], hf, s * 32), b2[s], c2);

        // finalize: cutoff scale, gather neighbor feature, scatter-add
#pragma unroll
        for (int v = 0; v < 8; v++) {
            int p = p0 + v + 8 * hf;
            float wij = (c2[v] + bias2) * rcut[p];
            int j = idx_j[p], i = idx_i[p];
            float contrib = wij * h[(size_t)j * FDIM + n];
            atomicAdd(&agg[(size_t)i * FDIM + n], contrib);
        }
    }
}

// ======================= out = ssp(agg@Wo1+bo1)@Wo2 + bo2 =======================
#define OUT_TILES 4
__global__ void __launch_bounds__(256) schnet_out(const float* agg, const __bf16* Wo1t,
                                                  const float* bo1, const __bf16* Wo2t,
                                                  const float* bo2, float* out) {
    __shared__ __align__(16) __bf16 tS[16][136];
    const int lane = threadIdx.x & 31;
    const int wave = threadIdx.x >> 5;
    const int hf   = lane >> 4;
    const int n    = wave * 16 + (lane & 15);
    const int m    = lane & 15;

    v16bf b1[4], b2[4];
#pragma unroll
    for (int s = 0; s < 4; s++) {
        b1[s] = load_bfrag(Wo1t, FDIM, n, hf, s * 32);
        b2[s] = load_bfrag(Wo2t, FDIM, n, hf, s * 32);
    }
    const float bias1 = bo1[n];
    const float bias2 = bo2[n];

    for (int t = 0; t < OUT_TILES; t++) {
        const int row0 = (blockIdx.x * OUT_TILES + t) * 16;
        const float* arow = agg + (size_t)(row0 + m) * FDIM;

        v8f c1 = vzero();
#pragma unroll
        for (int s = 0; s < 4; s++)
            c1 = wmma_bf16(load_afrag_f32(arow, hf, s * 32), b1[s], c1);

        __syncthreads();
#pragma unroll
        for (int v = 0; v < 8; v++)
            tS[v + 8 * hf][n] = (__bf16)sspf(c1[v] + bias1);
        __syncthreads();

        v8f c2 = vzero();
#pragma unroll
        for (int s = 0; s < 4; s++)
            c2 = wmma_bf16(load_afrag_lds(&tS[m][0], hf, s * 32), b2[s], c2);

#pragma unroll
        for (int v = 0; v < 8; v++) {
            int rm = v + 8 * hf;
            out[(size_t)(row0 + rm) * FDIM + n] = c2[v] + bias2;
        }
    }
}

extern "C" void kernel_launch(void* const* d_in, const int* in_sizes, int n_in,
                              void* d_out, int out_size, void* d_ws, size_t ws_size,
                              hipStream_t stream) {
    const float* x     = (const float*)d_in[0];
    const float* f_ij  = (const float*)d_in[1];
    const int*   idx_i = (const int*)d_in[2];
    const int*   idx_j = (const int*)d_in[3];
    const float* rcut  = (const float*)d_in[4];
    const float* W_in  = (const float*)d_in[5];
    const float* b_in  = (const float*)d_in[6];
    const float* Wf1   = (const float*)d_in[7];
    const float* bf1   = (const float*)d_in[8];
    const float* Wf2   = (const float*)d_in[9];
    const float* bf2   = (const float*)d_in[10];
    const float* Wo1   = (const float*)d_in[11];
    const float* bo1   = (const float*)d_in[12];
    const float* Wo2   = (const float*)d_in[13];
    const float* bo2   = (const float*)d_in[14];
    float* out = (float*)d_out;

    // workspace layout
    char* ws = (char*)d_ws;
    float*  agg   = (float*)(ws + 0);                       // 4,096,000 B
    float*  h     = (float*)(ws + 4096000);                 // 4,096,000 B
    __bf16* Wt_in = (__bf16*)(ws + 8192000);                // 32,768 B
    __bf16* Wf2t  = (__bf16*)(ws + 8224768);                // 32,768 B
    __bf16* Wo1t  = (__bf16*)(ws + 8257536);                // 32,768 B
    __bf16* Wo2t  = (__bf16*)(ws + 8290304);                // 32,768 B
    __bf16* Wf1t  = (__bf16*)(ws + 8323072);                // 8,192 B

    schnet_prep<<<256, 256, 0, stream>>>(W_in, Wf1, Wf2, Wo1, Wo2,
                                         agg, Wt_in, Wf2t, Wo1t, Wo2t, Wf1t);
    schnet_atom_in<<<BN / 16 / AT_TILES, 256, 0, stream>>>(x, Wt_in, b_in, h);
    schnet_pair<<<NP / 16 / PAIR_TILES, 256, 0, stream>>>(f_ij, idx_i, idx_j, rcut,
                                                          Wf1t, bf1, Wf2t, bf2, h, agg);
    schnet_out<<<BN / 16 / OUT_TILES, 256, 0, stream>>>(agg, Wo1t, bo1, Wo2t, bo2, out);
}
